// GraphEncoder_Attn_new_7825430413999
// MI455X (gfx1250) — matrix operand order repaired and checked by reading, not compiled, not run on hardware
//
#include <hip/hip_runtime.h>
#include <math.h>

typedef __attribute__((ext_vector_type(16))) __bf16 v16bf;
typedef __attribute__((ext_vector_type(8)))  __bf16 v8bf;
typedef __attribute__((ext_vector_type(8)))  float  v8f;

#define SHUF16(lo, hi) __builtin_shufflevector(lo, hi, 0,1,2,3,4,5,6,7,8,9,10,11,12,13,14,15)

// ---------------- helpers ----------------
__device__ __forceinline__ unsigned short f2bf_bits(float f) {
  unsigned u = __float_as_uint(f);
  u += 0x7FFFu + ((u >> 16) & 1u);           // round-to-nearest-even
  return (unsigned short)(u >> 16);
}
// total-order encoding of float into uint for atomicMax
__device__ __forceinline__ unsigned fkey(float f) {
  unsigned u = __float_as_uint(f);
  return (u & 0x80000000u) ? ~u : (u | 0x80000000u);
}
__device__ __forceinline__ float fdec(unsigned k) {
  unsigned u = (k & 0x80000000u) ? (k & 0x7FFFFFFFu) : ~k;
  return __uint_as_float(u);
}

// ---------------- WMMA GEMM ----------------
// C[M,Nout] = A[M,K](bf16,row-major) * Bt[Nout,K](bf16, "W^T" row-major) with
// f32 accumulation via v_wmma_f32_16x16x32_bf16. Each wave: 16 rows x 64 cols.
// Full-width blocks take a branch-free software-pipelined path: the steady-state
// loop prefetches K+32 unconditionally (last iteration peeled), so loads overlap
// the WMMAs instead of stalling on s_wait_loadcnt 0.
__global__ __launch_bounds__(256) void gemm_bf16_wmma(
    const __bf16* __restrict__ A, int lda,
    const __bf16* __restrict__ Bt, int ldb,
    float* __restrict__ Cf, unsigned short* __restrict__ Cb, int ldc,
    int Mtiles, int Ntiles, int Kdim,
    const float* __restrict__ bias, float scale)
{
  const int lane = threadIdx.x;                  // 0..31
  const int mt = blockIdx.x * 8 + threadIdx.y;   // wave-uniform
  if (mt >= Mtiles) return;                      // whole-wave exit (EXEC stays all-1s)
  const int nt0  = blockIdx.y * 4;
  const int hi   = (lane & 16) ? 1 : 0;
  const int kbA  = hi * 16;                      // A: lanes 16-31 hold K+16..31
  const int kbB  = hi * 16;                      // B: lanes 16-31 hold K+16..31
  const int ncol = lane & 15;
  const int mrow = mt * 16 + ncol;               // A row for this lane (M = lane&15)
  int nsub = Ntiles - nt0; if (nsub > 4) nsub = 4;

  v8f z = {0.f,0.f,0.f,0.f,0.f,0.f,0.f,0.f};
  v8f acc[4] = {z, z, z, z};

  const __bf16* arow = A + (long long)mrow * lda + kbA;
  const __bf16* brow[4];
#pragma unroll
  for (int s = 0; s < 4; ++s) {
    int sc = (s < nsub) ? s : 0;                 // clamp so tail path has valid ptrs
    brow[s] = Bt + (long long)((nt0 + sc) * 16 + ncol) * ldb + kbB;
  }

  if (nsub == 4) {
    // ---- branch-free double-buffered fast path, last iteration peeled ----
    v8bf al, ah, bl[4], bh[4];
    al = *(const v8bf*)(arow);
    ah = *(const v8bf*)(arow + 8);
#pragma unroll
    for (int s = 0; s < 4; ++s) {
      bl[s] = *(const v8bf*)(brow[s]);
      bh[s] = *(const v8bf*)(brow[s] + 8);
    }
    int k = 0;
    for (; k + 32 < Kdim; k += 32) {
      // unconditional prefetch of next K-step (k+32 < Kdim guaranteed)
      v8bf al2 = *(const v8bf*)(arow + k + 32);
      v8bf ah2 = *(const v8bf*)(arow + k + 40);
      v8bf bl2[4], bh2[4];
#pragma unroll
      for (int s = 0; s < 4; ++s) {
        bl2[s] = *(const v8bf*)(brow[s] + k + 32);
        bh2[s] = *(const v8bf*)(brow[s] + k + 40);
      }
      v16bf av = SHUF16(al, ah);
#pragma unroll
      for (int s = 0; s < 4; ++s) {
        v16bf bv = SHUF16(bl[s], bh[s]);
        acc[s] = __builtin_amdgcn_wmma_f32_16x16x32_bf16(
            false, av, false, bv, (short)0, acc[s], false, false);
      }
      al = al2; ah = ah2;
#pragma unroll
      for (int s = 0; s < 4; ++s) { bl[s] = bl2[s]; bh[s] = bh2[s]; }
    }
    // peeled final K-step
    {
      v16bf av = SHUF16(al, ah);
#pragma unroll
      for (int s = 0; s < 4; ++s) {
        v16bf bv = SHUF16(bl[s], bh[s]);
        acc[s] = __builtin_amdgcn_wmma_f32_16x16x32_bf16(
            false, av, false, bv, (short)0, acc[s], false, false);
      }
    }
  } else {
    // ---- ragged-tail path (rare: last grid.y block of scores GEMM) ----
    for (int k = 0; k < Kdim; k += 32) {
      v8bf alo = *(const v8bf*)(arow + k);
      v8bf ahi = *(const v8bf*)(arow + k + 8);
      v16bf av = SHUF16(alo, ahi);
#pragma unroll
      for (int s = 0; s < 4; ++s) {
        if (s >= nsub) break;                    // wave-uniform
        v8bf blo = *(const v8bf*)(brow[s] + k);
        v8bf bhi = *(const v8bf*)(brow[s] + k + 8);
        v16bf bv = SHUF16(blo, bhi);
        acc[s] = __builtin_amdgcn_wmma_f32_16x16x32_bf16(
            false, av, false, bv, (short)0, acc[s], false, false);
      }
    }
  }

#pragma unroll
  for (int s = 0; s < 4; ++s) {
    if (s >= nsub) break;
    int n = (nt0 + s) * 16 + ncol;
    float bb = bias ? bias[n] : 0.f;
#pragma unroll
    for (int v = 0; v < 8; ++v) {                // D: VGPR v -> M = v + hi*8
      int m = mt * 16 + v + hi * 8;
      float val = acc[s][v] * scale + bb;
      if (Cf) Cf[(long long)m * ldc + n] = val;
      if (Cb) Cb[(long long)m * ldc + n] = f2bf_bits(val);
    }
  }
}

// ---------------- elementwise / graph kernels ----------------
__global__ void k_zero32(unsigned* __restrict__ p, long long n) {
  long long t = (long long)blockIdx.x * blockDim.x + threadIdx.x;
  if (t < n) p[t] = 0u;
}

__global__ void k_transpose_bf(const float* __restrict__ W, unsigned short* __restrict__ Wt,
                               int K, int Nout) {
  long long t = (long long)blockIdx.x * blockDim.x + threadIdx.x;
  if (t >= (long long)K * Nout) return;
  int k = (int)(t / Nout), n = (int)(t % Nout);
  Wt[(long long)n * K + k] = f2bf_bits(W[t]);
}

__global__ void k_convert_bf(const float* __restrict__ in, unsigned short* __restrict__ out,
                             long long total) {
  long long t = (long long)blockIdx.x * blockDim.x + threadIdx.x;
  if (t < total) out[t] = f2bf_bits(in[t]);
}

// x[n, f*8+j] = node_feats[n, idx[n,f], j]  (bf16 out)
__global__ void k_gather(const float* __restrict__ nf, const int* __restrict__ idxb,
                         unsigned short* __restrict__ xg, int N) {
  int t = blockIdx.x * blockDim.x + threadIdx.x;
  if (t >= N * 8) return;
  int n = t >> 3, f = t & 7;
  int d = idxb[t];
  const float* p = nf + ((long long)n * 32 + d) * 8;
  unsigned short* o = xg + (long long)n * 64 + f * 8;
#pragma unroll
  for (int j = 0; j < 8; ++j) o[j] = f2bf_bits(p[j]);
}

// e_src/e_dst [N,H] from h [N,H*C]
__global__ void k_node_scores(const float* __restrict__ hbuf, const float* __restrict__ as,
                              const float* __restrict__ ad, float* __restrict__ esrc,
                              float* __restrict__ edst, int N, int H, int C) {
  int t = blockIdx.x * blockDim.x + threadIdx.x;
  if (t >= N * H) return;
  int n = t / H, h = t % H;
  const float* hp = hbuf + (long long)n * H * C + (long long)h * C;
  const float* ap = as + h * C;
  const float* dp = ad + h * C;
  float s1 = 0.f, s2 = 0.f;
  for (int c = 0; c < C; ++c) { float x = hp[c]; s1 += x * ap[c]; s2 += x * dp[c]; }
  esrc[t] = s1; edst[t] = s2;
}

__global__ void k_edge_max(const int* __restrict__ ei, int E, int Ep,
                           const float* __restrict__ esrc, const float* __restrict__ edst,
                           unsigned* __restrict__ mbits, int H) {
  long long t = (long long)blockIdx.x * blockDim.x + threadIdx.x;
  if (t >= (long long)Ep * H) return;
  int eid = (int)(t / H), h = (int)(t % H);
  int s, d;
  if (eid < E) { s = ei[eid]; d = ei[E + eid]; } else { s = d = eid - E; }
  float e = esrc[s * H + h] + edst[d * H + h];
  e = e > 0.f ? e : 0.2f * e;                    // leaky_relu(0.2)
  atomicMax(&mbits[d * H + h], fkey(e));
}

__global__ void k_edge_exp(const int* __restrict__ ei, int E, int Ep,
                           const float* __restrict__ esrc, const float* __restrict__ edst,
                           const unsigned* __restrict__ mbits, float* __restrict__ denom,
                           float* __restrict__ alpha, int H) {
  long long t = (long long)blockIdx.x * blockDim.x + threadIdx.x;
  if (t >= (long long)Ep * H) return;
  int eid = (int)(t / H), h = (int)(t % H);
  int s, d;
  if (eid < E) { s = ei[eid]; d = ei[E + eid]; } else { s = d = eid - E; }
  float e = esrc[s * H + h] + edst[d * H + h];
  e = e > 0.f ? e : 0.2f * e;
  float p = expf(e - fdec(mbits[d * H + h]));
  alpha[t] = p;
  atomicAdd(&denom[d * H + h], p);
}

__global__ void k_alpha_div(const int* __restrict__ ei, int E, int Ep,
                            const float* __restrict__ denom, float* __restrict__ alpha, int H) {
  long long t = (long long)blockIdx.x * blockDim.x + threadIdx.x;
  if (t >= (long long)Ep * H) return;
  int eid = (int)(t / H), h = (int)(t % H);
  int d = (eid < E) ? ei[E + eid] : (eid - E);
  alpha[t] = alpha[t] / (denom[d * H + h] + 1e-16f);
}

// agg[dst, j] += h[src, j] * alpha[e, j/C]
__global__ void k_scatter(const int* __restrict__ ei, int E, int Ep,
                          const float* __restrict__ hbuf, const float* __restrict__ alpha,
                          float* __restrict__ agg, int H, int C) {
  int HCl = H * C;
  long long t = (long long)blockIdx.x * blockDim.x + threadIdx.x;
  if (t >= (long long)Ep * HCl) return;
  int eid = (int)(t / HCl);
  int j = (int)(t % HCl);
  int h = j / C;
  int s, d;
  if (eid < E) { s = ei[eid]; d = ei[E + eid]; } else { s = d = eid - E; }
  float v = hbuf[(long long)s * HCl + j] * alpha[(long long)eid * H + h];
  atomicAdd(&agg[(long long)d * HCl + j], v);
}

// out = elu(agg + bias) -> bf16
__global__ void k_post_elu(const float* __restrict__ agg, const float* __restrict__ bias,
                           unsigned short* __restrict__ out, long long total, int HCl) {
  long long t = (long long)blockIdx.x * blockDim.x + threadIdx.x;
  if (t >= total) return;
  int j = (int)(t % HCl);
  float v = agg[t] + bias[j];
  v = v > 0.f ? v : (expf(v) - 1.f);
  out[t] = f2bf_bits(v);
}

// layer-3: mean over heads + bias, mask; emit row-major and transposed bf16
__global__ void k_post_mean(const float* __restrict__ agg, const float* __restrict__ b3,
                            const int* __restrict__ maskb, int N, int C, int H, int Npad,
                            unsigned short* __restrict__ xg, unsigned short* __restrict__ xgt) {
  int t = blockIdx.x * blockDim.x + threadIdx.x;
  if (t >= N * C) return;
  int n = t / C, c = t % C;
  float s = 0.f;
  for (int h = 0; h < H; ++h) s += agg[(long long)n * H * C + (long long)h * C + c];
  float v = (s / (float)H + b3[c]) * (float)maskb[n];
  unsigned short bv = f2bf_bits(v);
  xg[t] = bv;
  xgt[(long long)c * Npad + n] = bv;
}

// row softmax over N, write bf16 weights padded with zeros to Npad
__global__ void k_softmax(const float* __restrict__ scores, int N, int Npad,
                          unsigned short* __restrict__ wb) {
  int row = blockIdx.x;
  const float* srow = scores + (long long)row * Npad;
  __shared__ float red[256];
  int tid = threadIdx.x;
  float mx = -3.4e38f;
  for (int n = tid; n < N; n += 256) mx = fmaxf(mx, srow[n]);
  red[tid] = mx; __syncthreads();
  for (int s = 128; s > 0; s >>= 1) { if (tid < s) red[tid] = fmaxf(red[tid], red[tid + s]); __syncthreads(); }
  float m = red[0]; __syncthreads();
  float sm = 0.f;
  for (int n = tid; n < N; n += 256) sm += expf(srow[n] - m);
  red[tid] = sm; __syncthreads();
  for (int s = 128; s > 0; s >>= 1) { if (tid < s) red[tid] += red[tid + s]; __syncthreads(); }
  float inv = 1.f / red[0];
  for (int n = tid; n < Npad; n += 256) {
    float v = (n < N) ? expf(srow[n] - m) * inv : 0.f;
    wb[(long long)row * Npad + n] = f2bf_bits(v);
  }
}

// ---------------- host orchestration ----------------
extern "C" void kernel_launch(void* const* d_in, const int* in_sizes, int n_in,
                              void* d_out, int out_size, void* d_ws, size_t ws_size,
                              hipStream_t stream) {
  const int*   mask       = (const int*)d_in[0];
  const int*   idx        = (const int*)d_in[1];
  const float* node_feats = (const float*)d_in[2];
  const int*   ei         = (const int*)d_in[3];
  const float* value_W    = (const float*)d_in[4];
  const float* value_b    = (const float*)d_in[5];
  const float* w1  = (const float*)d_in[6];
  const float* as1 = (const float*)d_in[7];
  const float* ad1 = (const float*)d_in[8];
  const float* b1  = (const float*)d_in[9];
  const float* w2  = (const float*)d_in[10];
  const float* as2 = (const float*)d_in[11];
  const float* ad2 = (const float*)d_in[12];
  const float* b2  = (const float*)d_in[13];
  const float* w3  = (const float*)d_in[14];
  const float* as3 = (const float*)d_in[15];
  const float* ad3 = (const float*)d_in[16];
  const float* b3  = (const float*)d_in[17];
  const float* tq  = (const float*)d_in[18];
  const float* attn_W = (const float*)d_in[19];
  const float* attn_b = (const float*)d_in[20];

  const int N  = in_sizes[2] / 256;          // node_feats [N,32,8]
  const int E  = in_sizes[3] / 2;
  const int B  = in_sizes[0] / N;
  const int S  = in_sizes[18] / 128;         // seq_len
  const int H = 8, Dd = 64, HC = 512, Dg = 128, HCg = 1024;
  const int Ep = E + N;
  const int Npad = (N + 31) & ~31;
  const int Mt = (N + 15) / 16;              // 625 for N=10000
  const int St = (S + 15) / 16;              // 6 for S=96

  char* base = (char*)d_ws;
  size_t off = 0;
  auto alloc = [&](size_t bytes) { size_t r = off; off = (off + bytes + 255) & ~(size_t)255; return r; };
  unsigned short* wv_t   = (unsigned short*)(base + alloc((size_t)Dd * Dd * 2));
  unsigned short* w1_t   = (unsigned short*)(base + alloc((size_t)HC * Dd * 2));
  unsigned short* w2_t   = (unsigned short*)(base + alloc((size_t)HC * HC * 2));
  unsigned short* w3_t   = (unsigned short*)(base + alloc((size_t)HCg * HC * 2));
  unsigned short* wa_t   = (unsigned short*)(base + alloc((size_t)Dg * Dg * 2));
  unsigned short* tq_bf  = (unsigned short*)(base + alloc((size_t)S * Dg * 2));
  unsigned short* xg_bf  = (unsigned short*)(base + alloc((size_t)N * Dd * 2));
  unsigned short* act_bf = (unsigned short*)(base + alloc((size_t)N * HC * 2));
  float*          h_f    = (float*)(base + alloc((size_t)N * HCg * 4));
  float*          agg    = (float*)(base + alloc((size_t)N * HCg * 4));
  float*          esrc   = (float*)(base + alloc((size_t)N * H * 4));
  float*          edst   = (float*)(base + alloc((size_t)N * H * 4));
  unsigned*       mbits  = (unsigned*)(base + alloc((size_t)N * H * 4));
  float*          denom  = (float*)(base + alloc((size_t)N * H * 4));
  float*          alpha  = (float*)(base + alloc((size_t)Ep * H * 4));
  unsigned short* xg128  = (unsigned short*)(base + alloc((size_t)N * Dg * 2));
  unsigned short* xgt_bf = (unsigned short*)(base + alloc((size_t)Dg * Npad * 2));
  unsigned short* k_bf   = (unsigned short*)(base + alloc((size_t)N * Dg * 2));
  float*          scores = (float*)(base + alloc((size_t)S * Npad * 4));
  unsigned short* w_bf   = (unsigned short*)(base + alloc((size_t)S * Npad * 2));
  (void)ws_size; (void)n_in; (void)out_size;

  auto nb = [](long long total) { return (unsigned)((total + 255) / 256); };
  auto gemm = [&](const void* A, int lda, const void* Bt, int ldb, float* Cf,
                  unsigned short* Cb, int ldc, int Mtl, int Ntl, int K,
                  const float* bias, float scale) {
    dim3 g((unsigned)((Mtl + 7) / 8), (unsigned)((Ntl + 3) / 4));
    dim3 blk(32, 8);
    gemm_bf16_wmma<<<g, blk, 0, stream>>>((const __bf16*)A, lda, (const __bf16*)Bt, ldb,
                                          Cf, Cb, ldc, Mtl, Ntl, K, bias, scale);
  };

  // one-time weight prep (deterministic; re-done every call)
  k_transpose_bf<<<nb((long long)Dd * Dd), 256, 0, stream>>>(value_W, wv_t, Dd, Dd);
  k_transpose_bf<<<nb((long long)Dd * HC), 256, 0, stream>>>(w1, w1_t, Dd, HC);
  k_transpose_bf<<<nb((long long)HC * HC), 256, 0, stream>>>(w2, w2_t, HC, HC);
  k_transpose_bf<<<nb((long long)HC * HCg), 256, 0, stream>>>(w3, w3_t, HC, HCg);
  k_transpose_bf<<<nb((long long)Dg * Dg), 256, 0, stream>>>(attn_W, wa_t, Dg, Dg);
  k_convert_bf<<<nb((long long)S * Dg), 256, 0, stream>>>(tq, tq_bf, (long long)S * Dg);

  auto gat_edges = [&](const float* as, const float* ad, int C) {
    int HCl = H * C;
    k_node_scores<<<nb((long long)N * H), 256, 0, stream>>>(h_f, as, ad, esrc, edst, N, H, C);
    k_zero32<<<nb((long long)N * H), 256, 0, stream>>>(mbits, (long long)N * H);
    k_zero32<<<nb((long long)N * H), 256, 0, stream>>>((unsigned*)denom, (long long)N * H);
    k_zero32<<<nb((long long)N * HCl), 256, 0, stream>>>((unsigned*)agg, (long long)N * HCl);
    k_edge_max<<<nb((long long)Ep * H), 256, 0, stream>>>(ei, E, Ep, esrc, edst, mbits, H);
    k_edge_exp<<<nb((long long)Ep * H), 256, 0, stream>>>(ei, E, Ep, esrc, edst, mbits, denom, alpha, H);
    k_alpha_div<<<nb((long long)Ep * H), 256, 0, stream>>>(ei, E, Ep, denom, alpha, H);
    k_scatter<<<nb((long long)Ep * HCl), 256, 0, stream>>>(ei, E, Ep, h_f, alpha, agg, H, C);
  };

  const float inv_scale = 1.0f / sqrtf((float)Dg);

  for (int b = 0; b < B; ++b) {
    // gather + value projection (bias, no activation) -> act_bf [N,64]
    k_gather<<<nb((long long)N * 8), 256, 0, stream>>>(node_feats, idx + (long long)b * N * 8, xg_bf, N);
    gemm(xg_bf, Dd, wv_t, Dd, nullptr, act_bf, Dd, Mt, Dd / 16, Dd, value_b, 1.f);

    // GAT layer 1 (K=64 -> 512), elu
    gemm(act_bf, Dd, w1_t, Dd, h_f, nullptr, HC, Mt, HC / 16, Dd, nullptr, 1.f);
    gat_edges(as1, ad1, Dd);
    k_post_elu<<<nb((long long)N * HC), 256, 0, stream>>>(agg, b1, act_bf, (long long)N * HC, HC);

    // GAT layer 2 (512 -> 512), elu
    gemm(act_bf, HC, w2_t, HC, h_f, nullptr, HC, Mt, HC / 16, HC, nullptr, 1.f);
    gat_edges(as2, ad2, Dd);
    k_post_elu<<<nb((long long)N * HC), 256, 0, stream>>>(agg, b2, act_bf, (long long)N * HC, HC);

    // GAT layer 3 (512 -> 1024), head mean + bias + mask
    gemm(act_bf, HC, w3_t, HC, h_f, nullptr, HCg, Mt, HCg / 16, HC, nullptr, 1.f);
    gat_edges(as3, ad3, Dg);
    k_zero32<<<nb((long long)Dg * Npad / 2), 256, 0, stream>>>((unsigned*)xgt_bf, (long long)Dg * Npad / 2);
    k_post_mean<<<nb((long long)N * Dg), 256, 0, stream>>>(agg, b3, mask + (long long)b * N, N, Dg, H, Npad, xg128, xgt_bf);

    // K = xG @ attn_W + attn_b  (bf16 out, row-major == Bt layout for scores GEMM)
    gemm(xg128, Dg, wa_t, Dg, nullptr, k_bf, Dg, Mt, Dg / 16, Dg, attn_b, 1.f);
    // scores = (tq @ K^T) / sqrt(Dg)  -> f32 [S, Npad(stride)]
    gemm(tq_bf, Dg, k_bf, Dg, scores, nullptr, Npad, St, Mt, Dg, nullptr, inv_scale);
    // softmax over nodes -> bf16 weights, zero pad
    k_softmax<<<S, 256, 0, stream>>>(scores, N, Npad, w_bf);
    // out = w @ xG  -> d_out[b]
    gemm(w_bf, Npad, xgt_bf, Npad, (float*)d_out + (long long)b * S * Dg, nullptr, Dg,
         St, Dg / 16, Npad, nullptr, 1.f);
  }
}
